// InjectedEncoder_46377056862592
// MI455X (gfx1250) — matrix operand order repaired and checked by reading, not compiled, not run on hardware
//
#include <hip/hip_runtime.h>
#include <hip/hip_bf16.h>

// ---------------------------------------------------------------------------
// CDNA5 / gfx1250 implementation.
//  - wave32 WMMA (v_wmma_f32_16x16x32_bf16) for all GEMM-shaped work
//  - GEMM fast paths: vectorized 16B fragment loads (A, W^T-style B) and
//    LDS-staged B tiles for row-major K x N activations; 2 N-tiles per wave
//  - fused conv_in + 4x4/stride4 downsample through LDS
//  - centroids = GEMM -> in-place softmax (f32 row rewritten as bf16 P row)
//    -> GEMM with transposed C strides (lands directly in NCHW carry)
// ---------------------------------------------------------------------------

typedef __bf16 bf16;
typedef __attribute__((ext_vector_type(16))) __bf16 v16bf;
typedef __attribute__((ext_vector_type(8)))  float  v8f;
typedef unsigned short u16;
typedef __attribute__((ext_vector_type(8))) u16 u16x8;

#define EPSF 1e-5f

static __device__ __forceinline__ v16bf mk16(u16x8 lo, u16x8 hi)
{
  union { u16x8 h[2]; v16bf v; } u;
  u.h[0] = lo; u.h[1] = hi;
  return u.v;
}

// ===========================================================================
// Generic strided bf16 GEMM with f32 accumulate via WMMA.
//   C[b][m,n] = (flags&1 ? C : 0) + sum_k A[b][m,k]*B[b][k,n]
//               + bias[b][ flags&4 ? n : m ]      (bias optional)
//   flags&2 -> ReLU
// One wave per block; each wave owns a 16(M) x 32(N) strip (two 16x16 WMMA
// tiles sharing the A fragment). Fragment layouts used here (same k order for A and B):
//   A 16x32 bf16 : lane&15 = row M, element i holds K = i + 16*(lane>=16)
//   B 32x16 bf16 : lane&15 = col N, element i holds K = i + 16*(lane>=16)
//   C 16x16 f32  : VGPR r -> row r + 8*(lane>=16), col = lane&15
// ===========================================================================
__global__ __launch_bounds__(32) void gemm_wmma_bf16(
    const bf16* __restrict__ A, long aB, long am, long ak,
    const bf16* __restrict__ B, long bB, long bk, long bn,
    float* __restrict__ C, long cB, long cm, long cn,
    const float* __restrict__ bias, long biasB,
    int M, int N, int K, int flags)
{
  __shared__ u16 btile[32 * 32];   // staged B tile (rows k x cols n)

  int b    = blockIdx.z;
  int tm   = blockIdx.y * 16;
  int tn0  = blockIdx.x * 32;
  int lane = threadIdx.x;
  int half = lane >> 4;
  int l15  = lane & 15;

  const bf16* Ab = A + (long)b * aB;
  const bf16* Bb = B + (long)b * bB;

  v8f acc0 = {}, acc1 = {};
  int arow = tm + l15;

  bool fast = (K % 32 == 0) && (tm + 16 <= M) && (tn0 + 32 <= N) && (ak == 1) &&
              (bk == 1 || bn == 1);

  if (fast) {
    const u16* Au = (const u16*)Ab;
    const u16* Bu = (const u16*)Bb;
    const u16* arp = Au + (long)arow * am + half * 16;  // A run base (k += kk)
    if (bk == 1) {
      // ---- B contiguous in K per column: direct vector loads ----
      const u16* bp0 = Bu + (long)(tn0 + l15) * bn + half * 16;
      const u16* bp1 = Bu + (long)(tn0 + 16 + l15) * bn + half * 16;
      for (int kk = 0; kk < K; kk += 32) {
        if (kk + 32 < K) __builtin_prefetch(arp + kk + 32, 0, 0);
        v16bf af = mk16(*(const u16x8*)(arp + kk), *(const u16x8*)(arp + kk + 8));
        v16bf b0 = mk16(*(const u16x8*)(bp0 + kk), *(const u16x8*)(bp0 + kk + 8));
        v16bf b1 = mk16(*(const u16x8*)(bp1 + kk), *(const u16x8*)(bp1 + kk + 8));
        acc0 = __builtin_amdgcn_wmma_f32_16x16x32_bf16(false, af, false, b0,
                                                       (short)0, acc0, false, false);
        acc1 = __builtin_amdgcn_wmma_f32_16x16x32_bf16(false, af, false, b1,
                                                       (short)0, acc1, false, false);
      }
    } else {
      // ---- bn==1: row-major K x N activation. Stage 32x32 tile in LDS ----
      for (int kk = 0; kk < K; kk += 32) {
        const u16* brow = Bu + (long)(kk + lane) * bk + tn0;  // 32 contiguous cols
        u16x8 r0 = *(const u16x8*)(brow + 0);
        u16x8 r1 = *(const u16x8*)(brow + 8);
        u16x8 r2 = *(const u16x8*)(brow + 16);
        u16x8 r3 = *(const u16x8*)(brow + 24);
        *(u16x8*)&btile[lane * 32 + 0]  = r0;
        *(u16x8*)&btile[lane * 32 + 8]  = r1;
        *(u16x8*)&btile[lane * 32 + 16] = r2;
        *(u16x8*)&btile[lane * 32 + 24] = r3;
        if (kk + 32 < K) __builtin_prefetch(arp + kk + 32, 0, 0);
        v16bf af = mk16(*(const u16x8*)(arp + kk), *(const u16x8*)(arp + kk + 8));
        v16bf b0, b1;
#pragma unroll
        for (int i = 0; i < 16; ++i) {
          int krow = half * 16 + i;
          b0[i] = __builtin_bit_cast(bf16, btile[krow * 32 + l15]);
          b1[i] = __builtin_bit_cast(bf16, btile[krow * 32 + 16 + l15]);
        }
        acc0 = __builtin_amdgcn_wmma_f32_16x16x32_bf16(false, af, false, b0,
                                                       (short)0, acc0, false, false);
        acc1 = __builtin_amdgcn_wmma_f32_16x16x32_bf16(false, af, false, b1,
                                                       (short)0, acc1, false, false);
      }
    }
  } else {
    // ---- generic bounds-checked path (tile edges, K % 32 != 0) ----
    bool arowOK = arow < M;
    for (int t2 = 0; t2 < 2; ++t2) {
      int  tn     = tn0 + 16 * t2;
      int  bcol   = tn + l15;
      bool bcolOK = bcol < N;
      v8f  acc    = {};
      for (int kk = 0; kk < K; kk += 32) {
        v16bf af, bfr;
#pragma unroll
        for (int j = 0; j < 8; ++j) {
          int k0 = kk + 2 * j + half * 16;
          int k1 = k0 + 1;
          af[2 * j]     = (arowOK && k0 < K) ? Ab[(long)arow * am + (long)k0 * ak] : (bf16)0.0f;
          af[2 * j + 1] = (arowOK && k1 < K) ? Ab[(long)arow * am + (long)k1 * ak] : (bf16)0.0f;
          int kb0 = kk + 2 * j + half * 16;
          int kb1 = kb0 + 1;
          bfr[2 * j]     = (bcolOK && kb0 < K) ? Bb[(long)kb0 * bk + (long)bcol * bn] : (bf16)0.0f;
          bfr[2 * j + 1] = (bcolOK && kb1 < K) ? Bb[(long)kb1 * bk + (long)bcol * bn] : (bf16)0.0f;
        }
        acc = __builtin_amdgcn_wmma_f32_16x16x32_bf16(false, af, false, bfr,
                                                      (short)0, acc, false, false);
      }
      if (t2 == 0) acc0 = acc; else acc1 = acc;
    }
  }

  // ---- epilogue ----
  float* Cb = C + (long)b * cB;
  const float* biasb = bias ? bias + (long)b * biasB : nullptr;
#pragma unroll
  for (int t2 = 0; t2 < 2; ++t2) {
    int tn = tn0 + 16 * t2;
    v8f acc = t2 ? acc1 : acc0;
#pragma unroll
    for (int r = 0; r < 8; ++r) {
      int row = tm + r + half * 8;
      int col = tn + l15;
      if (row < M && col < N) {
        float v = acc[r];
        if (biasb) v += biasb[(flags & 4) ? col : row];
        long idx = (long)row * cm + (long)col * cn;
        if (flags & 1) v += Cb[idx];
        if (flags & 2) v = fmaxf(v, 0.0f);
        Cb[idx] = v;
      }
    }
  }
}

// ===========================================================================
// f32 -> bf16 conversion (grid-stride)
// ===========================================================================
__global__ void cvt_f32_bf16(const float* __restrict__ in, bf16* __restrict__ out, long n)
{
  long i = blockIdx.x * (long)blockDim.x + threadIdx.x;
  long s = (long)gridDim.x * blockDim.x;
  for (; i < n; i += s) out[i] = (bf16)in[i];
}

// ===========================================================================
// Fused conv_in(3x3,pad1,3->64) + downsample(4x4,stride4,64->64).
// Block = (b, oy, ox-group-of-4). h0 patch (64ch x 4y x 16x) staged in LDS.
// ===========================================================================
__global__ __launch_bounds__(256) void conv_in_ds(
    const float* __restrict__ x, const float* __restrict__ in_w,
    const float* __restrict__ in_b, const float* __restrict__ ds_w,
    const float* __restrict__ ds_b, float* __restrict__ h1)
{
  __shared__ float lds[64 * 64]; // [cm][ry*16+rx]
  int oxg = blockIdx.x;  // 0..7
  int oy  = blockIdx.y;  // 0..31
  int b   = blockIdx.z;  // 0..31
  int t   = threadIdx.x;
  const float* xb = x + (long)b * 3 * 128 * 128;

#pragma unroll 1
  for (int i = 0; i < 16; ++i) {
    int v   = t * 16 + i;
    int cm  = v >> 6;
    int rem = v & 63;
    int ry  = rem >> 4, rx = rem & 15;
    int y = 4 * oy + ry, xx = 16 * oxg + rx;
    float acc = in_b[cm];
    for (int c = 0; c < 3; ++c)
      for (int dy = 0; dy < 3; ++dy)
        for (int dx = 0; dx < 3; ++dx) {
          int iy = y + dy - 1, ix = xx + dx - 1;
          if (iy >= 0 && iy < 128 && ix >= 0 && ix < 128)
            acc += in_w[((cm * 3 + c) * 3 + dy) * 3 + dx] * xb[(c * 128 + iy) * 128 + ix];
        }
    lds[v] = acc;
  }
  __syncthreads();

  int co = t & 63, px = t >> 6;
  float acc = ds_b[co];
  for (int cm = 0; cm < 64; ++cm)
    for (int ky = 0; ky < 4; ++ky)
      for (int kx = 0; kx < 4; ++kx)
        acc += ds_w[((co * 64 + cm) * 4 + ky) * 4 + kx] * lds[cm * 64 + ky * 16 + 4 * px + kx];
  int ox = 4 * oxg + px;
  h1[(((long)b * 64 + co) * 32 + oy) * 32 + ox] = acc;
}

// ===========================================================================
// Direct 3x3 conv (64->64, 32x32, pad1), optional ReLU
// ===========================================================================
__global__ void conv3x3_relu(const float* __restrict__ in, const float* __restrict__ w,
                             const float* __restrict__ bias, float* __restrict__ out, int relu)
{
  long total = 32L * 64 * 1024;
  for (long idx = blockIdx.x * (long)blockDim.x + threadIdx.x; idx < total;
       idx += (long)gridDim.x * blockDim.x) {
    int p = (int)(idx & 1023), co = (int)((idx >> 10) & 63), b = (int)(idx >> 16);
    int y = p >> 5, xc = p & 31;
    float acc = bias[co];
    const float* ib = in + (long)b * 64 * 1024;
    for (int ci = 0; ci < 64; ++ci) {
      const float* ic = ib + ci * 1024;
      const float* wc = w + (co * 64 + ci) * 9;
      for (int dy = 0; dy < 3; ++dy) {
        int iy = y + dy - 1;
        if ((unsigned)iy >= 32u) continue;
        for (int dx = 0; dx < 3; ++dx) {
          int ix = xc + dx - 1;
          if ((unsigned)ix >= 32u) continue;
          acc += wc[dy * 3 + dx] * ic[iy * 32 + ix];
        }
      }
    }
    out[idx] = relu ? fmaxf(acc, 0.0f) : acc;
  }
}

// h2 = rb_bs + 1x1(h1, rb_ws) + rb_b1 + 3x3(t, rb_w1)
__global__ void rb_combine(const float* __restrict__ h1, const float* __restrict__ tt,
                           const float* __restrict__ ws, const float* __restrict__ bs,
                           const float* __restrict__ w1, const float* __restrict__ b1,
                           float* __restrict__ h2)
{
  long total = 32L * 64 * 1024;
  for (long idx = blockIdx.x * (long)blockDim.x + threadIdx.x; idx < total;
       idx += (long)gridDim.x * blockDim.x) {
    int p = (int)(idx & 1023), co = (int)((idx >> 10) & 63), b = (int)(idx >> 16);
    int y = p >> 5, xc = p & 31;
    const float* hb = h1 + (long)b * 64 * 1024;
    const float* tb = tt + (long)b * 64 * 1024;
    float acc = bs[co] + b1[co];
    for (int ci = 0; ci < 64; ++ci) {
      acc += ws[co * 64 + ci] * hb[ci * 1024 + p];
      const float* wc = w1 + (co * 64 + ci) * 9;
      const float* tc = tb + ci * 1024;
      for (int dy = 0; dy < 3; ++dy) {
        int iy = y + dy - 1;
        if ((unsigned)iy >= 32u) continue;
        for (int dx = 0; dx < 3; ++dx) {
          int ix = xc + dx - 1;
          if ((unsigned)ix >= 32u) continue;
          acc += wc[dy * 3 + dx] * tc[iy * 32 + ix];
        }
      }
    }
    h2[idx] = acc;
  }
}

// ===========================================================================
// Instance-norm of a (32,64,32,32) tensor -> bf16.
// transposed=1 : out[(b*1024+p)*64 + c]  (pixel-major, for WMMA A fragments)
// transposed=0 : out[(b*64+c)*1024 + p]  (channel-major, for WMMA B operand)
// ===========================================================================
__global__ __launch_bounds__(256) void inorm_to_bf16(const float* __restrict__ in,
                                                     bf16* __restrict__ out, int transposed)
{
  __shared__ float red[256], red2[256];
  int c = blockIdx.x, b = blockIdx.y, t = threadIdx.x;
  const float* ib = in + ((long)b * 64 + c) * 1024;
  float v[4], s = 0.f, s2 = 0.f;
  for (int i = 0; i < 4; ++i) {
    float xv = ib[t + 256 * i];
    v[i] = xv; s += xv; s2 += xv * xv;
  }
  red[t] = s; red2[t] = s2; __syncthreads();
  for (int o = 128; o > 0; o >>= 1) {
    if (t < o) { red[t] += red[t + o]; red2[t] += red2[t + o]; }
    __syncthreads();
  }
  float mean = red[0] * (1.f / 1024.f);
  float var  = red2[0] * (1.f / 1024.f) - mean * mean;
  float rs   = rsqrtf(var + EPSF);
  for (int i = 0; i < 4; ++i) {
    int p = t + 256 * i;
    bf16 o = (bf16)((v[i] - mean) * rs);
    if (transposed) out[((long)b * 1024 + p) * 64 + c] = o;
    else            out[((long)b * 64 + c) * 1024 + p] = o;
  }
}

// ===========================================================================
// z = inorm(sin_sobel(carry)) written bf16 in (b, 192, 1024) layout.
// blockIdx.x = out channel 0..191 (0-63 identity, 64-127 sobel-x, 128-191 sobel-y)
// ===========================================================================
__global__ __launch_bounds__(256) void sobel_inorm(const float* __restrict__ carry,
                                                   bf16* __restrict__ z)
{
  __shared__ float red[256], red2[256];
  int c = blockIdx.x, b = blockIdx.y, t = threadIdx.x;
  int mode = c >> 6;      // 0 identity, 1 gx, 2 gy
  int cs   = c & 63;
  const float* in = carry + ((long)b * 64 + cs) * 1024;
  float v[4], s = 0.f, s2 = 0.f;
  for (int i = 0; i < 4; ++i) {
    int p = t + 256 * i;
    float val;
    if (mode == 0) {
      val = in[p];
    } else {
      int y = p >> 5, xc = p & 31;
      float acc = 0.f;
      for (int dy = -2; dy <= 2; ++dy)
        for (int dx = -2; dx <= 2; ++dx) {
          if (dy == 0 && dx == 0) continue;
          int iy = y + dy, ix = xc + dx;
          if ((unsigned)iy >= 32u || (unsigned)ix >= 32u) continue;
          float d   = (float)(dy * dy + dx * dx);
          float wgt = (mode == 1 ? (float)dx : (float)dy) / d;
          acc += wgt * in[iy * 32 + ix];
        }
      val = acc;
    }
    v[i] = val; s += val; s2 += val * val;
  }
  red[t] = s; red2[t] = s2; __syncthreads();
  for (int o = 128; o > 0; o >>= 1) {
    if (t < o) { red[t] += red[t + o]; red2[t] += red2[t + o]; }
    __syncthreads();
  }
  float mean = red[0] * (1.f / 1024.f);
  float var  = red2[0] * (1.f / 1024.f) - mean * mean;
  float rs   = rsqrtf(var + EPSF);
  bf16* zb = z + ((long)b * 192 + c) * 1024;
  for (int i = 0; i < 4; ++i) zb[t + 256 * i] = (bf16)((v[i] - mean) * rs);
}

// ===========================================================================
// |cent_n|^2 per centroid
// ===========================================================================
__global__ void centroid_norms(const float* __restrict__ cent, float* __restrict__ cn)
{
  int n = blockIdx.x * blockDim.x + threadIdx.x;
  if (n < 1024) {
    float s = 0.f;
    for (int k = 0; k < 64; ++k) { float c = cent[n * 64 + k]; s += c * c; }
    cn[n] = s;
  }
}

// ===========================================================================
// softmax over centroids of (2*S - |c|^2); row rewritten in place as bf16 P.
// ===========================================================================
__global__ __launch_bounds__(256) void softmax_centroids(float* __restrict__ logits,
                                                         const float* __restrict__ cn)
{
  __shared__ float red[256];
  int p = blockIdx.x, b = blockIdx.y, t = threadIdx.x;
  float* row = logits + ((long)b * 1024 + p) * 1024;
  float v[4], mx = -3.4e38f;
  for (int i = 0; i < 4; ++i) {
    int n = t + 256 * i;
    v[i] = 2.f * row[n] - cn[n];
    mx = fmaxf(mx, v[i]);
  }
  red[t] = mx; __syncthreads();
  for (int o = 128; o > 0; o >>= 1) { if (t < o) red[t] = fmaxf(red[t], red[t + o]); __syncthreads(); }
  mx = red[0]; __syncthreads();
  float s = 0.f;
  for (int i = 0; i < 4; ++i) { v[i] = __expf(v[i] - mx); s += v[i]; }
  red[t] = s; __syncthreads();
  for (int o = 128; o > 0; o >>= 1) { if (t < o) red[t] += red[t + o]; __syncthreads(); }
  float inv = 1.f / red[0];
  __syncthreads();
  bf16* prow = (bf16*)row;               // reuse row storage (2048-elt bf16 stride)
  for (int i = 0; i < 4; ++i) prow[t + 256 * i] = (bf16)(v[i] * inv);
}

// ===========================================================================
// carry += clip(leak) * y[:,0:64] * sigmoid(y[:,64:128])
// ===========================================================================
__global__ void gate_update(float* __restrict__ carry, const float* __restrict__ y,
                            const float* __restrict__ leak)
{
  float lk = fminf(fmaxf(leak[0], 1e-3f), 1e3f);
  long total = 32L * 64 * 1024;
  for (long idx = blockIdx.x * (long)blockDim.x + threadIdx.x; idx < total;
       idx += (long)gridDim.x * blockDim.x) {
    int p = (int)(idx & 1023), c = (int)((idx >> 10) & 63), b = (int)(idx >> 16);
    const float* yb = y + (long)b * 128 * 1024;
    float a = yb[c * 1024 + p];
    float g = yb[(c + 64) * 1024 + p];
    carry[idx] += lk * a * (1.f / (1.f + __expf(-g)));
  }
}

// ===========================================================================
// Assemble cs (32 x 5184) from oc (32 x 193 x 32 x 32)
// ===========================================================================
__global__ void build_cs(const float* __restrict__ oc, float* __restrict__ cs)
{
  long total = 32L * 5184;
  for (long idx = blockIdx.x * (long)blockDim.x + threadIdx.x; idx < total;
       idx += (long)gridDim.x * blockDim.x) {
    int b = (int)(idx / 5184), j = (int)(idx % 5184);
    const float* ob = oc + (long)b * 193 * 1024;
    float v;
    if (j < 64) {
      const float* r = ob + j * 1024;
      float s = 0.f; for (int p = 0; p < 1024; ++p) s += r[p];
      v = s * (1.f / 1024.f);
    } else if (j < 2112) {
      int jj = j - 64; int c = 64 + (jj >> 5); int h = jj & 31;
      const float* r = ob + c * 1024 + h * 32;
      float s = 0.f; for (int w = 0; w < 32; ++w) s += r[w];
      v = s * (1.f / 32.f);
    } else if (j < 4160) {
      int jj = j - 2112; int c = 128 + (jj >> 5); int w = jj & 31;
      const float* r = ob + c * 1024 + w;
      float s = 0.f; for (int h = 0; h < 32; ++h) s += r[h * 32];
      v = s * (1.f / 32.f);
    } else {
      v = ob[192 * 1024 + (j - 4160)];
    }
    cs[idx] = v;
  }
}

// ===========================================================================
// Host-side orchestration
// ===========================================================================
extern "C" void kernel_launch(void* const* d_in, const int* in_sizes, int n_in,
                              void* d_out, int out_size, void* d_ws, size_t ws_size,
                              hipStream_t stream)
{
  (void)in_sizes; (void)n_in; (void)out_size; (void)ws_size;
  const float* x       = (const float*)d_in[0];
  const float* inj_lat = (const float*)d_in[1];
  const float* in_w    = (const float*)d_in[2];
  const float* in_b    = (const float*)d_in[3];
  const float* ds_w    = (const float*)d_in[4];
  const float* ds_b    = (const float*)d_in[5];
  const float* rb_w0   = (const float*)d_in[6];
  const float* rb_b0   = (const float*)d_in[7];
  const float* rb_w1   = (const float*)d_in[8];
  const float* rb_b1   = (const float*)d_in[9];
  const float* rb_ws   = (const float*)d_in[10];
  const float* rb_bs   = (const float*)d_in[11];
  const float* cent    = (const float*)d_in[12];
  const float* hyp_w   = (const float*)d_in[13];
  const float* hyp_b   = (const float*)d_in[14];
  const float* leak    = (const float*)d_in[15];
  const float* orb_w0  = (const float*)d_in[16];
  const float* orb_b0  = (const float*)d_in[17];
  const float* orb_w1  = (const float*)d_in[18];
  const float* orb_b1  = (const float*)d_in[19];
  const float* orb_ws  = (const float*)d_in[20];
  const float* orb_bs  = (const float*)d_in[21];
  const float* l1_wi   = (const float*)d_in[22];
  const float* l1_bi   = (const float*)d_in[23];
  const float* l1_wo   = (const float*)d_in[24];
  const float* l1_bo   = (const float*)d_in[25];
  const float* l1_ws   = (const float*)d_in[26];
  const float* l1_bs   = (const float*)d_in[27];
  const float* l2_wi   = (const float*)d_in[28];
  const float* l2_bi   = (const float*)d_in[29];
  const float* l2_wo   = (const float*)d_in[30];
  const float* l2_bo   = (const float*)d_in[31];
  const float* l2_ws   = (const float*)d_in[32];
  const float* l2_bs   = (const float*)d_in[33];
  const float* fl_w    = (const float*)d_in[34];
  const float* fl_b    = (const float*)d_in[35];
  float* out = (float*)d_out;

  // ---- workspace bump allocator (256B aligned) ----
  char*  wsb = (char*)d_ws;
  size_t off = 0;
  auto alloc = [&](size_t bytes) -> void* {
    off = (off + 255) & ~(size_t)255;
    void* p = wsb + off;
    off += bytes;
    return p;
  };
  const long P = 45376;
  float* h1      = (float*)alloc(32L * 64 * 1024 * 4);
  float* tconv   = (float*)alloc(32L * 64 * 1024 * 4);
  float* h2      = (float*)alloc(32L * 64 * 1024 * 4);
  bf16*  fT      = (bf16*) alloc(32L * 1024 * 64 * 2);
  bf16*  cbf     = (bf16*) alloc(1024L * 64 * 2);
  float* cnrm    = (float*)alloc(1024L * 4);
  float* logits  = (float*)alloc(32L * 1024 * 1024 * 4);   // also holds bf16 P in place
  float* carry   = (float*)alloc(32L * 64 * 1024 * 4);
  float* pbuf    = (float*)alloc(32L * P * 4);
  bf16*  pbf     = (bf16*) alloc(32L * P * 2);
  bf16*  injbf   = (bf16*) alloc(32L * 512 * 2);
  bf16*  hypwbf  = (bf16*) alloc(P * 512L * 2);
  bf16*  zbf     = (bf16*) alloc(32L * 192 * 1024 * 2);
  float* hid     = (float*)alloc(32L * 64 * 1024 * 4);
  bf16*  hidbf   = (bf16*) alloc(32L * 64 * 1024 * 2);
  float* ybuf    = (float*)alloc(32L * 128 * 1024 * 4);
  bf16*  oobf    = (bf16*) alloc(32L * 64 * 1024 * 2);
  bf16*  ow0bf   = (bf16*) alloc(193L * 64 * 2);
  bf16*  ow1bf   = (bf16*) alloc(193L * 193 * 2);
  bf16*  owsbf   = (bf16*) alloc(193L * 64 * 2);
  float* hid2    = (float*)alloc(32L * 193 * 1024 * 4);
  bf16*  hid2bf  = (bf16*) alloc(32L * 193 * 1024 * 2);
  float* oc      = (float*)alloc(32L * 193 * 1024 * 4);
  float* csb     = (float*)alloc(32L * 5184 * 4);
  bf16*  csbf    = (bf16*) alloc(32L * 5184 * 2);
  bf16*  l1wibf  = (bf16*) alloc(1024L * 5184 * 2);
  bf16*  l1wobf  = (bf16*) alloc(512L * 1024 * 2);
  bf16*  l1wsbf  = (bf16*) alloc(512L * 5184 * 2);
  bf16*  l2wibf  = (bf16*) alloc(512L * 512 * 2);
  bf16*  l2wobf  = (bf16*) alloc(512L * 512 * 2);
  bf16*  l2wsbf  = (bf16*) alloc(512L * 512 * 2);
  bf16*  flwbf   = (bf16*) alloc(512L * 512 * 2);
  float* t1      = (float*)alloc(32L * 1024 * 4);
  bf16*  t1bf    = (bf16*) alloc(32L * 1024 * 2);
  float* lat     = (float*)alloc(32L * 512 * 4);
  bf16*  latbf   = (bf16*) alloc(32L * 512 * 2);
  float* lat2    = (float*)alloc(32L * 512 * 4);
  bf16*  lat2bf  = (bf16*) alloc(32L * 512 * 2);

  auto cvt = [&](const float* in, bf16* o, long n) {
    long blocks = (n + 255) / 256;
    if (blocks > 4096) blocks = 4096;
    cvt_f32_bf16<<<dim3((unsigned)blocks), dim3(256), 0, stream>>>(in, o, n);
  };
  auto gemm = [&](const bf16* A, long aB, long am, long ak,
                  const bf16* B, long bB, long bk, long bn,
                  float* C, long cB, long cm, long cn,
                  const float* bias, long biasB,
                  int M, int N, int K, int flags, int batches) {
    dim3 g((N + 31) / 32, (M + 15) / 16, batches);
    gemm_wmma_bf16<<<g, dim3(32), 0, stream>>>(A, aB, am, ak, B, bB, bk, bn,
                                               C, cB, cm, cn, bias, biasB, M, N, K, flags);
  };

  // ---- convert static weights to bf16 ----
  cvt(cent, cbf, 1024L * 64);
  cvt(hyp_w, hypwbf, P * 512L);
  cvt(inj_lat, injbf, 32L * 512);
  cvt(orb_w0, ow0bf, 193L * 64);
  cvt(orb_w1, ow1bf, 193L * 193);
  cvt(orb_ws, owsbf, 193L * 64);
  cvt(l1_wi, l1wibf, 1024L * 5184);
  cvt(l1_wo, l1wobf, 512L * 1024);
  cvt(l1_ws, l1wsbf, 512L * 5184);
  cvt(l2_wi, l2wibf, 512L * 512);
  cvt(l2_wo, l2wobf, 512L * 512);
  cvt(l2_ws, l2wsbf, 512L * 512);
  cvt(fl_w, flwbf, 512L * 512);
  centroid_norms<<<dim3(4), dim3(256), 0, stream>>>(cent, cnrm);

  // ---- front convs ----
  conv_in_ds<<<dim3(8, 32, 32), dim3(256), 0, stream>>>(x, in_w, in_b, ds_w, ds_b, h1);
  conv3x3_relu<<<dim3(4096), dim3(256), 0, stream>>>(h1, rb_w0, rb_b0, tconv, 1);
  rb_combine<<<dim3(4096), dim3(256), 0, stream>>>(h1, tconv, rb_ws, rb_bs, rb_w1, rb_b1, h2);

  // ---- centroids: inorm -> S = f @ cent^T -> softmax -> P @ cent (into NCHW carry) ----
  inorm_to_bf16<<<dim3(64, 32), dim3(256), 0, stream>>>(h2, fT, 1);
  gemm(fT, 1024L * 64, 64, 1,             // A: pixels x dims
       cbf, 0, 1, 64,                     // B: cent^T (k=dim, n=centroid) -> bk==1 fast path
       logits, 1024L * 1024, 1024, 1,
       nullptr, 0, 1024, 1024, 64, 0, 32);
  softmax_centroids<<<dim3(1024, 32), dim3(256), 0, stream>>>(logits, cnrm);
  gemm((const bf16*)logits, 1024L * 2048, 2048, 1,  // A: P bf16, row stride 2048
       cbf, 0, 64, 1,                               // B: cent (k=centroid, n=dim) -> LDS path
       carry, 64L * 1024, 1, 1024,                  // C transposed -> NCHW
       nullptr, 0, 1024, 64, 1024, 0, 32);

  // ---- hypernetwork: p = inj @ hyp_w^T + hyp_b ----
  gemm(injbf, 0, 512, 1,
       hypwbf, 0, 1, 512,                 // B = hyp_w^T -> bk==1 fast path
       pbuf, 0, P, 1,
       hyp_b, 0, 32, (int)P, 512, 4, 1);
  cvt(pbuf, pbf, 32L * P);

  // per-batch hyper weight offsets inside p
  const long OFF_WIN = 0, OFF_BIN = 12288, OFF_WOUT = 12352,
             OFF_BOUT = 20544, OFF_WSK = 20672, OFF_BSK = 45248;

  // ---- 16 recurrent steps ----
  for (int step = 0; step < 16; ++step) {
    sobel_inorm<<<dim3(192, 32), dim3(256), 0, stream>>>(carry, zbf);
    // hid = relu(w_in @ z + b_in)
    gemm(pbf + OFF_WIN, P, 192, 1,
         zbf, 192L * 1024, 1024, 1,
         hid, 64L * 1024, 1024, 1,
         pbuf + OFF_BIN, P, 64, 1024, 192, 2, 32);
    cvt(hid, hidbf, 32L * 64 * 1024);
    // y = w_out @ hid + b_out
    gemm(pbf + OFF_WOUT, P, 64, 1,
         hidbf, 64L * 1024, 1024, 1,
         ybuf, 128L * 1024, 1024, 1,
         pbuf + OFF_BOUT, P, 128, 1024, 64, 0, 32);
    // y += w_sk @ z + b_sk
    gemm(pbf + OFF_WSK, P, 192, 1,
         zbf, 192L * 1024, 1024, 1,
         ybuf, 128L * 1024, 1024, 1,
         pbuf + OFF_BSK, P, 128, 1024, 192, 1, 32);
    gate_update<<<dim3(4096), dim3(256), 0, stream>>>(carry, ybuf, leak);
  }

  // ---- orbit head ----
  inorm_to_bf16<<<dim3(64, 32), dim3(256), 0, stream>>>(carry, oobf, 0);
  gemm(ow0bf, 0, 64, 1,
       oobf, 64L * 1024, 1024, 1,
       hid2, 193L * 1024, 1024, 1,
       orb_b0, 0, 193, 1024, 64, 2, 32);
  cvt(hid2, hid2bf, 32L * 193 * 1024);
  gemm(owsbf, 0, 64, 1,
       oobf, 64L * 1024, 1024, 1,
       oc, 193L * 1024, 1024, 1,
       orb_bs, 0, 193, 1024, 64, 0, 32);
  gemm(ow1bf, 0, 193, 1,
       hid2bf, 193L * 1024, 1024, 1,
       oc, 193L * 1024, 1024, 1,
       orb_b1, 0, 193, 1024, 193, 1, 32);

  // ---- cs assembly + latent MLPs ----
  build_cs<<<dim3(648), dim3(256), 0, stream>>>(oc, csb);
  cvt(csb, csbf, 32L * 5184);

  // lrb1: t1 = relu(cs@l1_wi^T+bi); lat = cs@l1_ws^T+bs; lat += t1@l1_wo^T+bo
  gemm(csbf, 0, 5184, 1, l1wibf, 0, 1, 5184, t1, 0, 1024, 1,
       l1_bi, 0, 32, 1024, 5184, 2 | 4, 1);
  cvt(t1, t1bf, 32L * 1024);
  gemm(csbf, 0, 5184, 1, l1wsbf, 0, 1, 5184, lat, 0, 512, 1,
       l1_bs, 0, 32, 512, 5184, 4, 1);
  gemm(t1bf, 0, 1024, 1, l1wobf, 0, 1, 1024, lat, 0, 512, 1,
       l1_bo, 0, 32, 512, 1024, 1 | 4, 1);
  cvt(lat, latbf, 32L * 512);

  // lrb2
  gemm(latbf, 0, 512, 1, l2wibf, 0, 1, 512, t1, 0, 512, 1,
       l2_bi, 0, 32, 512, 512, 2 | 4, 1);
  cvt(t1, t1bf, 32L * 512);
  gemm(latbf, 0, 512, 1, l2wsbf, 0, 1, 512, lat2, 0, 512, 1,
       l2_bs, 0, 32, 512, 512, 4, 1);
  gemm(t1bf, 0, 512, 1, l2wobf, 0, 1, 512, lat2, 0, 512, 1,
       l2_bo, 0, 32, 512, 512, 1 | 4, 1);
  cvt(lat2, lat2bf, 32L * 512);

  // final: out = lat2 @ fl_w^T + fl_b
  gemm(lat2bf, 0, 512, 1, flwbf, 0, 1, 512, out, 0, 512, 1,
       fl_b, 0, 32, 512, 512, 4, 1);
}
